// LaminarNet_56453050138732
// MI455X (gfx1250) — matrix operand order repaired and checked by reading, not compiled, not run on hardware
//
#include <hip/hip_runtime.h>
#include <hip/hip_bf16.h>
#include <stddef.h>

// ---------------------------------------------------------------------------
// LaminarNet for MI455X (gfx1250, wave32, WMMA, async-to-LDS).
// B=4, N=8192, D=1024, H=16, d_head=64, CHUNK=256, nc=32, 3D=3072.
// ---------------------------------------------------------------------------

#define BB   4
#define NN   8192
#define DD   1024
#define HH   16
#define DH   64
#define NCH  32        // chunks per sequence
#define CHK  256       // chunk length
#define D3   3072
#define BN   (BB * NN)         // 32768 token rows
#define BND  ((size_t)BN * DD) // 33,554,432

typedef __attribute__((ext_vector_type(16))) __bf16 v16bf;
typedef __attribute__((ext_vector_type(8)))  float  v8f;
typedef __attribute__((ext_vector_type(4)))  unsigned int u32x4; // POD 16B vector

union Frag16 { u32x4 u[2]; v16bf v; };   // 32 bytes = 16 bf16 = one WMMA operand

__device__ __forceinline__ float clampf(float x, float lo, float hi) {
  return fminf(fmaxf(x, lo), hi);
}

// gfx1250 async global->LDS copy (ASYNCcnt-tracked, VGLOBAL opcode 98).
// VDST = VGPR holding the 32-bit LDS byte address (low 32 bits of the generic
// LDS pointer == LDS offset per the aperture rules), VADDR = 64-bit global VA.
__device__ __forceinline__ void async_copy_b128(const void* g, void* l) {
  asm volatile("global_load_async_to_lds_b128 %0, %1, off"
               :
               : "v"((unsigned)(unsigned long long)l),
                 "v"((unsigned long long)g)
               : "memory");
}
__device__ __forceinline__ void wait_async0() {
  asm volatile("s_wait_asynccnt 0x0" ::: "memory");
}

// ---------------------------------------------------------------------------
// 1) RMSNorm: one block per token row of 1024.
// ---------------------------------------------------------------------------
__global__ __launch_bounds__(256)
void rmsnorm_kernel(const float* __restrict__ x, const float* __restrict__ scale,
                    float* __restrict__ xn) {
  __shared__ float red[8];
  const int t = threadIdx.x;
  const size_t row = blockIdx.x;
  const float* xr = x + row * DD;
  float vals[4];
  float ss = 0.f;
#pragma unroll
  for (int i = 0; i < 4; ++i) {
    vals[i] = xr[t + i * 256];
    ss += vals[i] * vals[i];
  }
  for (int off = 16; off > 0; off >>= 1) ss += __shfl_down(ss, off, 32);
  if ((t & 31) == 0) red[t >> 5] = ss;
  __syncthreads();
  if (t < 8) {
    float s = red[t];
    for (int off = 4; off > 0; off >>= 1) s += __shfl_down(s, off, 8);
    if (t == 0) red[0] = s;
  }
  __syncthreads();
  const float rms = rsqrtf(red[0] * (1.0f / (float)DD) + 1e-6f);
  float* o = xn + row * DD;
#pragma unroll
  for (int i = 0; i < 4; ++i) {
    float v = scale[t + i * 256] * vals[i] * rms;
    o[t + i * 256] = clampf(v, -60000.f, 60000.f);
  }
}

// ---------------------------------------------------------------------------
// 2) fp32 -> bf16 weight conversion.
// ---------------------------------------------------------------------------
__global__ __launch_bounds__(256)
void cvt_bf16_kernel(const float* __restrict__ src, __bf16* __restrict__ dst, int n) {
  int i = blockIdx.x * 256 + threadIdx.x;
  if (i < n) dst[i] = (__bf16)src[i];
}

// ---------------------------------------------------------------------------
// 3) Depthwise causal conv (K=4) + SiLU, emit bf16 GEMM-A operand.
// ---------------------------------------------------------------------------
__global__ __launch_bounds__(256)
void conv_silu_kernel(const float* __restrict__ xn, const float* __restrict__ w,
                      const float* __restrict__ bias, __bf16* __restrict__ xc) {
  const size_t i = (size_t)blockIdx.x * 256 + threadIdx.x;  // over B*N*D
  const int d = (int)(i & (DD - 1));
  const size_t bn = i >> 10;
  const int n = (int)(bn & (NN - 1));
  float acc = bias[d];
#pragma unroll
  for (int j = 0; j < 4; ++j) {
    const int nn = n - 3 + j;
    const float xv = (nn >= 0) ? xn[(ptrdiff_t)i + (ptrdiff_t)(j - 3) * DD] : 0.f;
    acc = fmaf(w[d * 4 + j], xv, acc);
  }
  const float s = acc / (1.f + __expf(-acc));  // SiLU
  xc[i] = (__bf16)s;
}

// ---------------------------------------------------------------------------
// 4) WMMA bf16 GEMM:  C[MxN] (+residual) = A[MxK] * Bm[KxN], fp32 accum.
//    Block tile 128x128, K-step 32, 8 waves (4 along M x 2 along N),
//    each wave: 2x4 grid of 16x16 v_wmma_f32_16x16x32_bf16 tiles.
//    Double-buffered LDS; A tile streamed with global_load_async_to_lds_b128
//    (no VGPR staging), B tile register-staged + transposed on ds_store so
//    both fragment loads are contiguous b128 LDS reads matching the ISA
//    operand layouts. One barrier per K-iteration.
// ---------------------------------------------------------------------------
#define TM 128
#define TN 128
#define TK 32
#define LSTR 40   // padded LDS row stride in bf16 (80B; 16B-aligned chunks)

__global__ __launch_bounds__(256)
void gemm_bf16_wmma_kernel(const __bf16* __restrict__ A, const __bf16* __restrict__ Bm,
                           float* __restrict__ C, const float* __restrict__ residual,
                           int M, int N, int K) {
  __shared__ __align__(16) __bf16 lds_a[2][TM * LSTR];
  __shared__ __align__(16) __bf16 lds_b[2][TN * LSTR];

  const int t = threadIdx.x;
  const int lane = t & 31;
  const int wv = t >> 5;          // 0..7
  const int wave_m = wv & 3;      // 4 waves along M  (each 32 rows)
  const int wave_n = wv >> 2;     // 2 waves along N  (each 64 cols)
  const int block_m = blockIdx.y * TM;
  const int block_n = blockIdx.x * TN;
  const int kh = lane >> 4;       // K-half select (ISA 16-bit operand layout)
  const int l15 = lane & 15;

  // Per-thread staging coordinates (two 8-element vectors each for A and B).
  const int ia0 = t * 8, ia1 = (t + 256) * 8;
  const int arow0 = ia0 >> 5, akk0 = ia0 & 31;   // A tile: [row][k]
  const int arow1 = ia1 >> 5, akk1 = ia1 & 31;
  const int bkk0 = ia0 >> 7, bnn0 = ia0 & 127;   // B tile: [k][n]
  const int bkk1 = ia1 >> 7, bnn1 = ia1 & 127;

  const __bf16* ag0 = A + (size_t)(block_m + arow0) * K + akk0;
  const __bf16* ag1 = A + (size_t)(block_m + arow1) * K + akk1;
  const __bf16* bg0 = Bm + (size_t)bkk0 * N + block_n + bnn0;
  const __bf16* bg1 = Bm + (size_t)bkk1 * N + block_n + bnn1;

  v8f acc[2][4];
#pragma unroll
  for (int i = 0; i < 2; ++i)
#pragma unroll
    for (int j = 0; j < 4; ++j) acc[i][j] = (v8f){};

  // ---- prologue: stage K-tile 0 into buffer 0 ----
  async_copy_b128(ag0, &lds_a[0][arow0 * LSTR + akk0]);
  async_copy_b128(ag1, &lds_a[0][arow1 * LSTR + akk1]);
  {
    const u32x4 rb0 = *(const u32x4*)bg0;
    const u32x4 rb1 = *(const u32x4*)bg1;
    const __bf16* e0 = (const __bf16*)&rb0;
    const __bf16* e1 = (const __bf16*)&rb1;
#pragma unroll
    for (int j = 0; j < 8; ++j) {
      lds_b[0][(bnn0 + j) * LSTR + bkk0] = e0[j];
      lds_b[0][(bnn1 + j) * LSTR + bkk1] = e1[j];
    }
  }
  wait_async0();
  __syncthreads();

  int p = 0;
  for (int k0 = 0; k0 < K; k0 += TK) {
    const bool nx = (k0 + TK) < K;  // uniform
    u32x4 rb0{}, rb1{};
    if (nx) {
      // kick off next A tile straight into the other LDS buffer (ASYNCcnt)
      async_copy_b128(ag0 + (k0 + TK), &lds_a[p ^ 1][arow0 * LSTR + akk0]);
      async_copy_b128(ag1 + (k0 + TK), &lds_a[p ^ 1][arow1 * LSTR + akk1]);
      // next B tile into registers (LOADcnt) while we compute below
      rb0 = *(const u32x4*)(bg0 + (size_t)(k0 + TK) * N);
      rb1 = *(const u32x4*)(bg1 + (size_t)(k0 + TK) * N);
    }

    // --- A fragments: lane l -> row M=l15; VGPR0-3 K=kh*8..+7, 4-7 K=16+kh*8..+7 ---
    Frag16 afr[2];
#pragma unroll
    for (int mt = 0; mt < 2; ++mt) {
      const int m = wave_m * 32 + mt * 16 + l15;
      afr[mt].u[0] = *(const u32x4*)(&lds_a[p][m * LSTR + kh * 8]);
      afr[mt].u[1] = *(const u32x4*)(&lds_a[p][m * LSTR + 16 + kh * 8]);
    }
    // --- B fragments: lane l -> col N=l15; VGPR0-3 K=kh*8..+7, 4-7 K=16+kh*8..+7 ---
    Frag16 bfr[4];
#pragma unroll
    for (int nt = 0; nt < 4; ++nt) {
      const int n = wave_n * 64 + nt * 16 + l15;
      const int kb = kh * 8;
      bfr[nt].u[0] = *(const u32x4*)(&lds_b[p][n * LSTR + kb]);
      bfr[nt].u[1] = *(const u32x4*)(&lds_b[p][n * LSTR + kb + 16]);
    }
#pragma unroll
    for (int mt = 0; mt < 2; ++mt)
#pragma unroll
      for (int nt = 0; nt < 4; ++nt)
        acc[mt][nt] = __builtin_amdgcn_wmma_f32_16x16x32_bf16(
            false, afr[mt].v, false, bfr[nt].v, (short)0, acc[mt][nt], false, false);

    if (nx) {
      // transpose-store next B tile (waits its global loads), drain async A
      const __bf16* e0 = (const __bf16*)&rb0;
      const __bf16* e1 = (const __bf16*)&rb1;
#pragma unroll
      for (int j = 0; j < 8; ++j) {
        lds_b[p ^ 1][(bnn0 + j) * LSTR + bkk0] = e0[j];
        lds_b[p ^ 1][(bnn1 + j) * LSTR + bkk1] = e1[j];
      }
      wait_async0();
    }
    __syncthreads();
    p ^= 1;
  }

  // --- epilogue: C layout VGPR r -> M = 8*(lane>=16) + r, N = lane&15 ---
#pragma unroll
  for (int mt = 0; mt < 2; ++mt) {
#pragma unroll
    for (int nt = 0; nt < 4; ++nt) {
      const int col = block_n + wave_n * 64 + nt * 16 + l15;
      const int rb = block_m + wave_m * 32 + mt * 16 + kh * 8;
#pragma unroll
      for (int r = 0; r < 8; ++r) {
        const size_t o = (size_t)(rb + r) * N + col;
        float v = acc[mt][nt][r];
        if (residual) v += residual[o];
        C[o] = v;
      }
    }
  }
}

// ---------------------------------------------------------------------------
// 5) Post-projection elementwise: dt=clip(softplus(dt_raw+bias)), gate=sigmoid,
//    RoPE on v, v_in = rope(v)*dt.  All updated in place inside `fused`.
//    One thread per rotation pair (b,n,h,i<32).
// ---------------------------------------------------------------------------
__device__ __forceinline__ float softplusf(float x) {
  return (x > 20.f) ? x : log1pf(__expf(x));
}

__global__ __launch_bounds__(256)
void post_in_kernel(float* __restrict__ fused, const float* __restrict__ dt_bias) {
  const size_t tid = (size_t)blockIdx.x * 256 + threadIdx.x;  // B*N*H*32 threads
  const int i = (int)(tid & 31);
  const size_t rem = tid >> 5;
  const int h = (int)(rem & (HH - 1));
  const size_t bn = rem >> 4;           // token row 0..32767
  const int n = (int)(bn & (NN - 1));
  const int d1 = h * DH + i, d2 = d1 + 32;
  float* base = fused + bn * D3;

  float dt1 = clampf(softplusf(base[d1] + dt_bias[d1]), 0.001f, 2.0f);
  float dt2 = clampf(softplusf(base[d2] + dt_bias[d2]), 0.001f, 2.0f);

  const float inv = __powf(10000.f, -(float)(2 * i) / (float)DH);
  float s, c;
  __sincosf((float)n * inv, &s, &c);
  const float v1 = base[DD + d1], v2 = base[DD + d2];

  base[d1] = dt1;
  base[d2] = dt2;
  base[DD + d1] = (v1 * c - v2 * s) * dt1;
  base[DD + d2] = (v1 * s + v2 * c) * dt2;
  base[2 * DD + d1] = 1.f / (1.f + __expf(-base[2 * DD + d1]));
  base[2 * DD + d2] = 1.f / (1.f + __expf(-base[2 * DD + d2]));
}

// ---------------------------------------------------------------------------
// 6a) Intra-chunk scan: thread = one channel of one (b,chunk). Two serial
//     passes of 256 steps (clip-of-prefix semantics like the reference).
// ---------------------------------------------------------------------------
__global__ __launch_bounds__(256)
void scan_chunk_kernel(const float* __restrict__ fused, float* __restrict__ L,
                       float* __restrict__ chunk_out, float* __restrict__ bdecay,
                       float* __restrict__ bout) {
  const int d = blockIdx.x * 256 + threadIdx.x;  // channel 0..1023
  const int bc = blockIdx.y;                     // 0..B*NC-1
  const int b = bc >> 5, c = bc & (NCH - 1);
  const size_t tok0 = (size_t)b * NN + (size_t)c * CHK;

  const float* dtp = fused + tok0 * D3 + d;
  float* Lp = L + tok0 * DD + d;
  float Lraw = 0.f, Lc = 0.f, Lmax = -1e30f;
  for (int i = 0; i < CHK; ++i) {
    Lraw += -dtp[(size_t)i * D3];
    Lc = clampf(Lraw, -20.f, 0.f);
    Lp[(size_t)i * DD] = Lc;
    Lmax = fmaxf(Lmax, Lc);
  }
  const float* vp = fused + tok0 * D3 + DD + d;
  float* cp = chunk_out + tok0 * DD + d;
  float cum = 0.f, co = 0.f;
  for (int i = 0; i < CHK; ++i) {
    const float Lst = Lp[(size_t)i * DD] - Lmax;
    cum = fmaf(__expf(-Lst), vp[(size_t)i * D3], cum);
    co = __expf(Lst) * cum;
    cp[(size_t)i * DD] = co;
  }
  const size_t bi = ((size_t)b * NCH + c) * DD + d;
  bdecay[bi] = Lc;
  bout[bi] = co;
}

// ---------------------------------------------------------------------------
// 6b) Inter-chunk carries: thread = one (b,d), 32 sequential chunks.
// ---------------------------------------------------------------------------
__global__ __launch_bounds__(256)
void scan_carry_kernel(const float* __restrict__ bdecay, const float* __restrict__ bout,
                       float* __restrict__ carries) {
  const int t = blockIdx.x * 256 + threadIdx.x;  // 0..4095
  const int b = t >> 10, d = t & (DD - 1);
  float raw = 0.f, stab = -1e30f;
  float cd[NCH];
  for (int c = 0; c < NCH; ++c) {
    raw += bdecay[((size_t)b * NCH + c) * DD + d];
    cd[c] = clampf(raw, -80.f, 0.f);
    stab = fmaxf(stab, cd[c]);
  }
  float run = 0.f;  // exclusive prefix sum of seeds
  for (int c = 0; c < NCH; ++c) {
    const float ncd = clampf(cd[c] - stab, -20.f, 0.f);
    carries[((size_t)b * NCH + c) * DD + d] = run * __expf(ncd);
    run = fmaf(bout[((size_t)b * NCH + c) * DD + d], __expf(-ncd), run);
  }
}

// ---------------------------------------------------------------------------
// 6c) Apply carries + head-mix (via LDS-staged row) + gate -> bf16 GEMM-A.
// ---------------------------------------------------------------------------
__global__ __launch_bounds__(256)
void mix_gate_kernel(const float* __restrict__ chunk_out, const float* __restrict__ L,
                     const float* __restrict__ carries, const float* __restrict__ fused,
                     const float* __restrict__ head_mix, __bf16* __restrict__ gact) {
  __shared__ float fbuf[DD];
  __shared__ float hm[HH * HH];
  const size_t bn = blockIdx.x;  // token row
  const int b = (int)(bn >> 13), n = (int)(bn & (NN - 1));
  const int c = n >> 8;
  hm[threadIdx.x] = head_mix[threadIdx.x];
  const size_t rowo = bn * DD;
#pragma unroll
  for (int i = 0; i < 4; ++i) {
    const int d = threadIdx.x + i * 256;
    const float carry = carries[((size_t)b * NCH + c) * DD + d];
    fbuf[d] = fmaf(carry, __expf(L[rowo + d]), chunk_out[rowo + d]);
  }
  __syncthreads();
  const float* grow = fused + bn * D3 + 2 * DD;
#pragma unroll
  for (int i = 0; i < 4; ++i) {
    const int dout = threadIdx.x + i * 256;
    const int m = dout >> 6, inner = dout & (DH - 1);
    float acc = 0.f;
#pragma unroll
    for (int h = 0; h < HH; ++h) acc = fmaf(fbuf[h * DH + inner], hm[h * HH + m], acc);
    gact[rowo + dout] = (__bf16)(acc * grow[dout]);
  }
}

// ---------------------------------------------------------------------------
// Launch sequence.
// ---------------------------------------------------------------------------
extern "C" void kernel_launch(void* const* d_in, const int* in_sizes, int n_in,
                              void* d_out, int out_size, void* d_ws, size_t ws_size,
                              hipStream_t stream) {
  (void)in_sizes; (void)n_in; (void)out_size; (void)ws_size;
  const float* x          = (const float*)d_in[0];
  const float* norm_scale = (const float*)d_in[1];
  const float* conv_w     = (const float*)d_in[2];
  const float* conv_b     = (const float*)d_in[3];
  const float* in_proj_w  = (const float*)d_in[4];
  const float* dt_bias    = (const float*)d_in[5];
  const float* head_mix   = (const float*)d_in[6];
  const float* out_proj_w = (const float*)d_in[7];
  float* out = (float*)d_out;

  // Workspace layout (buffers reused across phases):
  char* ws = (char*)d_ws;
  size_t off = 0;
  float*  xn_L      = (float*)(ws + off);  off += BND * 4;            // xn, later L
  __bf16* xc_gact   = (__bf16*)(ws + off); off += BND * 2;            // xc, later gated act
  float*  fused     = (float*)(ws + off);  off += (size_t)BN * D3 * 4;
  float*  chunk_out = (float*)(ws + off);  off += BND * 4;
  __bf16* wInB      = (__bf16*)(ws + off); off += (size_t)DD * D3 * 2;
  __bf16* wOutB     = (__bf16*)(ws + off); off += (size_t)DD * DD * 2;
  float*  bdecay    = (float*)(ws + off);  off += (size_t)BB * NCH * DD * 4;
  float*  bout      = (float*)(ws + off);  off += (size_t)BB * NCH * DD * 4;
  float*  carries   = (float*)(ws + off);  off += (size_t)BB * NCH * DD * 4;

  // 1) RMSNorm
  rmsnorm_kernel<<<BN, 256, 0, stream>>>(x, norm_scale, xn_L);
  // 2) weight conversion
  cvt_bf16_kernel<<<(DD * D3 + 255) / 256, 256, 0, stream>>>(in_proj_w, wInB, DD * D3);
  cvt_bf16_kernel<<<(DD * DD + 255) / 256, 256, 0, stream>>>(out_proj_w, wOutB, DD * DD);
  // 3) conv + SiLU -> bf16
  conv_silu_kernel<<<(unsigned)(BND / 256), 256, 0, stream>>>(xn_L, conv_w, conv_b, xc_gact);
  // 4) in_proj GEMM (WMMA): [32768x1024]*[1024x3072]
  {
    dim3 g(D3 / TN, BN / TM);
    gemm_bf16_wmma_kernel<<<g, 256, 0, stream>>>(xc_gact, wInB, fused, nullptr, BN, D3, DD);
  }
  // 5) dt / gate / RoPE / v*dt (in place)
  post_in_kernel<<<(unsigned)((size_t)BN * DD / 2 / 256), 256, 0, stream>>>(fused, dt_bias);
  // 6) scan
  {
    dim3 g(DD / 256, BB * NCH);
    scan_chunk_kernel<<<g, 256, 0, stream>>>(fused, xn_L, chunk_out, bdecay, bout);
  }
  scan_carry_kernel<<<(BB * DD) / 256, 256, 0, stream>>>(bdecay, bout, carries);
  mix_gate_kernel<<<BN, 256, 0, stream>>>(chunk_out, xn_L, carries, fused, head_mix, xc_gact);
  // 7) out_proj GEMM (WMMA) + residual: [32768x1024]*[1024x1024] + x
  {
    dim3 g(DD / TN, BN / TM);
    gemm_bf16_wmma_kernel<<<g, 256, 0, stream>>>(xc_gact, wOutB, out, x, BN, DD, DD);
  }
}
